// SimpleNet_37512244364142
// MI455X (gfx1250) — matrix operand order repaired and checked by reading, not compiled, not run on hardware
//
#include <hip/hip_runtime.h>
#include <hip/hip_bf16.h>
#include <math.h>

#define HDIM 64

typedef __attribute__((ext_vector_type(16))) _Float16 v16h;
typedef __attribute__((ext_vector_type(8)))  float    v8f;

// ---------------------------------------------------------------------------
// WMMA helpers (CDNA5 V_WMMA_F32_16X16X32_F16)
// ---------------------------------------------------------------------------
__device__ __forceinline__ v8f wmma16(v16h a, v16h b, v8f c) {
  return __builtin_amdgcn_wmma_f32_16x16x32_f16(false, a, false, b, (short)0, c,
                                                false, false);
}

// K index of element t (0..15) of a 16x32 f16 A-fragment: lane<16 holds K=0..7
// and 16..23; lane>=16 holds K=8..15 and 24..31 (hs = lane>>4).
__device__ __forceinline__ int a_k(int kc, int hs, int t) {
  return kc * 32 + ((t >> 3) << 4) + (hs << 3) + (t & 7);
}

// B-fragment (32x16 slice of a row-major [K][64] f16 matrix in LDS).
__device__ __forceinline__ v16h bfrag_lds(const _Float16* w, int k0, int n0,
                                          int lane) {
  const int n  = n0 + (lane & 15);
  const int ks = k0 + ((lane >> 4) << 3);
  v16h b;
#pragma unroll
  for (int i = 0; i < 16; ++i) b[i] = w[(ks + (i & 7) + ((i >> 3) << 4)) * HDIM + n];
  return b;
}

// Same, gathered from a row-major f32 matrix in global memory.
__device__ __forceinline__ v16h bfrag_glb(const float* __restrict__ w, int k0,
                                          int n0, int lane) {
  const int n  = n0 + (lane & 15);
  const int ks = k0 + ((lane >> 4) << 3);
  v16h b;
#pragma unroll
  for (int i = 0; i < 16; ++i) b[i] = (_Float16)w[(ks + (i & 7) + ((i >> 3) << 4)) * HDIM + n];
  return b;
}

// Load 8 contiguous f32 (two b128 loads), scale, pack into half of a v16h.
__device__ __forceinline__ void load8(const float* p, float s, v16h& a,
                                      int off) {
  const float4 u = *(const float4*)p;
  const float4 v = *(const float4*)(p + 4);
  a[off + 0] = (_Float16)(u.x * s); a[off + 1] = (_Float16)(u.y * s);
  a[off + 2] = (_Float16)(u.z * s); a[off + 3] = (_Float16)(u.w * s);
  a[off + 4] = (_Float16)(v.x * s); a[off + 5] = (_Float16)(v.y * s);
  a[off + 6] = (_Float16)(v.z * s); a[off + 7] = (_Float16)(v.w * s);
}

// ---------------------------------------------------------------------------
// Node encoder: x = relu(feat @ W1 + b1) @ W2 + b2   (feat: [N,2])
// ---------------------------------------------------------------------------
__global__ void __launch_bounds__(256, 1)
encoder_kernel(const float* __restrict__ feat, int N,
               const float* __restrict__ W1, const float* __restrict__ b1,
               const float* __restrict__ W2, const float* __restrict__ b2,
               float* __restrict__ xout) {
  __shared__ float sW2[HDIM * HDIM];
  __shared__ float hbuf[4][HDIM];
  const int tid = threadIdx.x;
  for (int i = tid; i < HDIM * HDIM; i += blockDim.x) sW2[i] = W2[i];
  __syncthreads();
  const int ch = tid & 63, g = tid >> 6;
  const int ngroups = (N + 3) >> 2;
  for (int gb = blockIdx.x; gb < ngroups; gb += gridDim.x) {
    const int node = gb * 4 + g;
    float f0 = 0.f, f1 = 0.f;
    if (node < N) { f0 = feat[(size_t)node * 2]; f1 = feat[(size_t)node * 2 + 1]; }
    hbuf[g][ch] = fmaxf(f0 * W1[ch] + f1 * W1[HDIM + ch] + b1[ch], 0.f);
    __syncthreads();
    float acc = b2[ch];
#pragma unroll 8
    for (int k = 0; k < HDIM; ++k) acc += hbuf[g][k] * sW2[k * HDIM + ch];
    if (node < N) xout[(size_t)node * HDIM + ch] = acc;
    __syncthreads();
  }
}

// ---------------------------------------------------------------------------
// Per-target in-degree counts
// ---------------------------------------------------------------------------
__global__ void __launch_bounds__(256, 1)
count_kernel(const int* __restrict__ ei, int E, float* __restrict__ cnt) {
  const int e = blockIdx.x * blockDim.x + threadIdx.x;
  if (e < E) atomicAdd(&cnt[ei[E + e]], 1.0f);
}

// ---------------------------------------------------------------------------
// Edge MLP + BN statistics (WMMA, 16 edges per wave-tile)
// ---------------------------------------------------------------------------
__global__ void __launch_bounds__(256, 1)
edge_stats_kernel(const float* __restrict__ eattr, int E,
                  const float* __restrict__ W1, const float* __restrict__ b1,
                  const float* __restrict__ W2, const float* __restrict__ b2,
                  float* __restrict__ stats) {
  __shared__ _Float16 sW2[HDIM * HDIM];
  __shared__ float sW1[HDIM], sB1[HDIM], sB2[HDIM];
  __shared__ float ssum[HDIM], ssq[HDIM];
  const int tid = threadIdx.x;
  for (int i = tid; i < HDIM * HDIM; i += blockDim.x) sW2[i] = (_Float16)W2[i];
  if (tid < HDIM) {
    sW1[tid] = W1[tid]; sB1[tid] = b1[tid]; sB2[tid] = b2[tid];
    ssum[tid] = 0.f; ssq[tid] = 0.f;
  }
  __syncthreads();
  const int lane = tid & 31;
  const int w = __builtin_amdgcn_readfirstlane(tid >> 5);  // scalar wave id
  const int hs = lane >> 4, r = lane & 15;
  v16h B[2][4];
#pragma unroll
  for (int kc = 0; kc < 2; ++kc)
#pragma unroll
    for (int nt = 0; nt < 4; ++nt)
      B[kc][nt] = bfrag_lds(sW2, kc * 32, nt * 16, lane);

  float accS[4] = {0.f, 0.f, 0.f, 0.f}, accQ[4] = {0.f, 0.f, 0.f, 0.f};
  const int ntiles = (E + 15) >> 4;
  const int nwaves = gridDim.x * 8;
  for (int t = blockIdx.x * 8 + w; t < ntiles; t += nwaves) {
    const int eb = t * 16;  // scalar
    if (eb + 16 <= E) {     // scalar branch: full tile, no guards
      const float ev = eattr[eb + r];
      v16h A[2];
#pragma unroll
      for (int kc = 0; kc < 2; ++kc) {
        v16h a;
#pragma unroll
        for (int tt = 0; tt < 16; ++tt) {
          const int ch = a_k(kc, hs, tt);
          a[tt] = (_Float16)fmaxf(ev * sW1[ch] + sB1[ch], 0.f);
        }
        A[kc] = a;
      }
#pragma unroll
      for (int nt = 0; nt < 4; ++nt) {
        v8f c = {};
        c = wmma16(A[0], B[0][nt], c);
        c = wmma16(A[1], B[1][nt], c);
        const float bb = sB2[nt * 16 + r];
#pragma unroll
        for (int j = 0; j < 8; ++j) {
          const float h = fmaxf(c[j] + bb, 0.f);
          accS[nt] += h; accQ[nt] += h * h;
        }
      }
    } else {                // tail tile (rare)
      const int er = eb + r;
      const float ev = (er < E) ? eattr[er] : 0.f;
      v16h A[2];
#pragma unroll
      for (int kc = 0; kc < 2; ++kc) {
        v16h a;
#pragma unroll
        for (int tt = 0; tt < 16; ++tt) {
          const int ch = a_k(kc, hs, tt);
          a[tt] = (_Float16)fmaxf(ev * sW1[ch] + sB1[ch], 0.f);
        }
        A[kc] = a;
      }
#pragma unroll
      for (int nt = 0; nt < 4; ++nt) {
        v8f c = {};
        c = wmma16(A[0], B[0][nt], c);
        c = wmma16(A[1], B[1][nt], c);
        const float bb = sB2[nt * 16 + r];
#pragma unroll
        for (int j = 0; j < 8; ++j) {
          if (eb + j + hs * 8 < E) {
            const float h = fmaxf(c[j] + bb, 0.f);
            accS[nt] += h; accQ[nt] += h * h;
          }
        }
      }
    }
  }
#pragma unroll
  for (int nt = 0; nt < 4; ++nt) {
    atomicAdd(&ssum[nt * 16 + r], accS[nt]);
    atomicAdd(&ssq[nt * 16 + r], accQ[nt]);
  }
  __syncthreads();
  if (tid < HDIM) {
    atomicAdd(&stats[tid], ssum[tid]);
    atomicAdd(&stats[HDIM + tid], ssq[tid]);
  }
}

__global__ void bn_finalize_kernel(const float* __restrict__ stats,
                                   const float* __restrict__ g,
                                   const float* __restrict__ bt, float En,
                                   float* __restrict__ scsh) {
  const int c = threadIdx.x;
  const float mu = stats[c] / En;
  const float var = stats[HDIM + c] / En - mu * mu;
  const float sc = g[c] * rsqrtf(var + 1e-5f);
  scsh[c] = sc;
  scsh[HDIM + c] = bt[c] - mu * sc;
}

// ---------------------------------------------------------------------------
// Recompute edge MLP, apply BN, msg = relu(src[ei0]+emb), scatter-add agg.
// ---------------------------------------------------------------------------
__global__ void __launch_bounds__(256, 1)
edge_message_kernel(const float* __restrict__ eattr,
                    const int* __restrict__ ei, int E,
                    const float* __restrict__ W1, const float* __restrict__ b1,
                    const float* __restrict__ W2, const float* __restrict__ b2,
                    const float* __restrict__ scsh,
                    const float* __restrict__ src, float* __restrict__ agg) {
  __shared__ _Float16 sW2[HDIM * HDIM];
  __shared__ float sW1[HDIM], sB1[HDIM], sB2[HDIM], sSc[HDIM], sSh[HDIM];
  const int tid = threadIdx.x;
  for (int i = tid; i < HDIM * HDIM; i += blockDim.x) sW2[i] = (_Float16)W2[i];
  if (tid < HDIM) {
    sW1[tid] = W1[tid]; sB1[tid] = b1[tid]; sB2[tid] = b2[tid];
    sSc[tid] = scsh[tid]; sSh[tid] = scsh[HDIM + tid];
  }
  __syncthreads();
  const int lane = tid & 31;
  const int w = __builtin_amdgcn_readfirstlane(tid >> 5);
  const int hs = lane >> 4, r = lane & 15;
  v16h B[2][4];
#pragma unroll
  for (int kc = 0; kc < 2; ++kc)
#pragma unroll
    for (int nt = 0; nt < 4; ++nt)
      B[kc][nt] = bfrag_lds(sW2, kc * 32, nt * 16, lane);

  const int ntiles = (E + 15) >> 4;
  const int nwaves = gridDim.x * 8;
  for (int t = blockIdx.x * 8 + w; t < ntiles; t += nwaves) {
    const int eb = t * 16;
    if (eb + 16 <= E) {  // full tile: unguarded
      const float ev = eattr[eb + r];
      v16h A[2];
#pragma unroll
      for (int kc = 0; kc < 2; ++kc) {
        v16h a;
#pragma unroll
        for (int tt = 0; tt < 16; ++tt) {
          const int ch = a_k(kc, hs, tt);
          a[tt] = (_Float16)fmaxf(ev * sW1[ch] + sB1[ch], 0.f);
        }
        A[kc] = a;
      }
      int si[8], ti[8];
#pragma unroll
      for (int j = 0; j < 8; ++j) {
        const int e = eb + j + hs * 8;
        si[j] = ei[e]; ti[j] = ei[E + e];
      }
#pragma unroll
      for (int nt = 0; nt < 4; ++nt) {
        v8f c = {};
        c = wmma16(A[0], B[0][nt], c);
        c = wmma16(A[1], B[1][nt], c);
        const int ch = nt * 16 + r;
        const float bb = sB2[ch], sc = sSc[ch], sh = sSh[ch];
#pragma unroll
        for (int j = 0; j < 8; ++j) {
          const float h = fmaxf(c[j] + bb, 0.f);
          const float emb = h * sc + sh;
          const float msg = fmaxf(src[(size_t)si[j] * HDIM + ch] + emb, 0.f);
          atomicAdd(&agg[(size_t)ti[j] * HDIM + ch], msg);
        }
      }
    } else {  // tail tile
      const int er = eb + r;
      const float ev = (er < E) ? eattr[er] : 0.f;
      v16h A[2];
#pragma unroll
      for (int kc = 0; kc < 2; ++kc) {
        v16h a;
#pragma unroll
        for (int tt = 0; tt < 16; ++tt) {
          const int ch = a_k(kc, hs, tt);
          a[tt] = (_Float16)fmaxf(ev * sW1[ch] + sB1[ch], 0.f);
        }
        A[kc] = a;
      }
      int si[8], ti[8];
#pragma unroll
      for (int j = 0; j < 8; ++j) {
        const int e = eb + j + hs * 8;
        if (e < E) { si[j] = ei[e]; ti[j] = ei[E + e]; }
        else       { si[j] = -1;    ti[j] = -1; }
      }
#pragma unroll
      for (int nt = 0; nt < 4; ++nt) {
        v8f c = {};
        c = wmma16(A[0], B[0][nt], c);
        c = wmma16(A[1], B[1][nt], c);
        const int ch = nt * 16 + r;
        const float bb = sB2[ch], sc = sSc[ch], sh = sSh[ch];
#pragma unroll
        for (int j = 0; j < 8; ++j) {
          if (si[j] >= 0) {
            const float h = fmaxf(c[j] + bb, 0.f);
            const float emb = h * sc + sh;
            const float msg = fmaxf(src[(size_t)si[j] * HDIM + ch] + emb, 0.f);
            atomicAdd(&agg[(size_t)ti[j] * HDIM + ch], msg);
          }
        }
      }
    }
  }
}

// ---------------------------------------------------------------------------
// out = relu(normalize((agg/cnt) @ lW + lb + tgt @ rW)), WMMA 16-node tiles.
// ---------------------------------------------------------------------------
__global__ void __launch_bounds__(256, 1)
node_update_kernel(const float* __restrict__ agg, const float* __restrict__ cnt,
                   const float* __restrict__ tgt, const float* __restrict__ lW,
                   const float* __restrict__ lb, const float* __restrict__ rW,
                   float* __restrict__ outp, int N) {
  __shared__ _Float16 sLW[HDIM * HDIM], sRW[HDIM * HDIM];
  __shared__ float sLB[HDIM];
  __shared__ float obuf[8][16][HDIM];
  __shared__ float rnorm[8][16];
  const int tid = threadIdx.x;
  for (int i = tid; i < HDIM * HDIM; i += blockDim.x) {
    sLW[i] = (_Float16)lW[i]; sRW[i] = (_Float16)rW[i];
  }
  if (tid < HDIM) sLB[tid] = lb[tid];
  __syncthreads();
  const int lane = tid & 31;
  const int w = __builtin_amdgcn_readfirstlane(tid >> 5);
  const int hs = lane >> 4, r = lane & 15;
  v16h BL[2][4], BR[2][4];
#pragma unroll
  for (int kc = 0; kc < 2; ++kc)
#pragma unroll
    for (int nt = 0; nt < 4; ++nt) {
      BL[kc][nt] = bfrag_lds(sLW, kc * 32, nt * 16, lane);
      BR[kc][nt] = bfrag_lds(sRW, kc * 32, nt * 16, lane);
    }
  const int ntiles = (N + 15) >> 4;
  const int nwaves = gridDim.x * 8;
  for (int t = blockIdx.x * 8 + w; t < ntiles; t += nwaves) {
    const int nb = t * 16;
    v16h Aa[2], At[2];
    if (nb + 16 <= N) {  // full tile: vector loads, no guards
      const int node = nb + r;
      const float inv = 1.f / fmaxf(cnt[node], 1.f);
      const float* pA = agg + (size_t)node * HDIM;
      const float* pT = tgt + (size_t)node * HDIM;
#pragma unroll
      for (int kc = 0; kc < 2; ++kc) {
        const int b0 = kc * 32 + hs * 8;
        v16h a, b;
        load8(pA + b0, inv, a, 0); load8(pA + b0 + 16, inv, a, 8);
        load8(pT + b0, 1.f, b, 0); load8(pT + b0 + 16, 1.f, b, 8);
        Aa[kc] = a; At[kc] = b;
      }
    } else {
      const int node = nb + r;
      const bool valid = node < N;
      float inv = 0.f;
      if (valid) inv = 1.f / fmaxf(cnt[node], 1.f);
#pragma unroll
      for (int kc = 0; kc < 2; ++kc) {
        v16h a, b;
#pragma unroll
        for (int tt = 0; tt < 16; ++tt) {
          const int ch = a_k(kc, hs, tt);
          float av = 0.f, tv = 0.f;
          if (valid) {
            av = agg[(size_t)node * HDIM + ch] * inv;
            tv = tgt[(size_t)node * HDIM + ch];
          }
          a[tt] = (_Float16)av; b[tt] = (_Float16)tv;
        }
        Aa[kc] = a; At[kc] = b;
      }
    }
#pragma unroll
    for (int nt = 0; nt < 4; ++nt) {
      v8f c = {};
      c = wmma16(Aa[0], BL[0][nt], c);
      c = wmma16(Aa[1], BL[1][nt], c);
      c = wmma16(At[0], BR[0][nt], c);
      c = wmma16(At[1], BR[1][nt], c);
      const int ch = nt * 16 + r;
      const float bb = sLB[ch];
#pragma unroll
      for (int j = 0; j < 8; ++j) obuf[w][j + hs * 8][ch] = c[j] + bb;
    }
    __builtin_amdgcn_wave_barrier();  // wave-private LDS, lockstep wave32
    if (lane < 16) {
      float s = 0.f;
#pragma unroll 8
      for (int k = 0; k < HDIM; ++k) { const float v = obuf[w][lane][k]; s += v * v; }
      rnorm[w][lane] = 1.f / fmaxf(sqrtf(s), 1e-12f);
    }
    __builtin_amdgcn_wave_barrier();
    if (nb + 16 <= N) {
      for (int i = lane; i < 16 * HDIM; i += 32) {
        const int m = i >> 6, ch = i & 63;
        outp[(size_t)(nb + m) * HDIM + ch] =
            fmaxf(obuf[w][m][ch] * rnorm[w][m], 0.f);
      }
    } else {
      for (int i = lane; i < 16 * HDIM; i += 32) {
        const int m = i >> 6, ch = i & 63;
        const int nd = nb + m;
        if (nd < N)
          outp[(size_t)nd * HDIM + ch] = fmaxf(obuf[w][m][ch] * rnorm[w][m], 0.f);
      }
    }
  }
}

// ---------------------------------------------------------------------------
// Final head: concat(5 x [Nv,64]) -> 320 -> 64 -> 64 -> 64 -> 2 -> log_softmax
// ---------------------------------------------------------------------------
__global__ void __launch_bounds__(128, 1)
final_mlp_kernel(const float* __restrict__ xvh, int Nv,
                 const float* __restrict__ W1, const float* __restrict__ b1,
                 const float* __restrict__ W2, const float* __restrict__ b2,
                 const float* __restrict__ W3, const float* __restrict__ b3,
                 const float* __restrict__ W4, const float* __restrict__ b4,
                 float* __restrict__ out) {
  __shared__ _Float16 s2[HDIM * HDIM], s3[HDIM * HDIM];
  __shared__ float obuf[4][16][HDIM];
  const int tid = threadIdx.x;
  for (int i = tid; i < HDIM * HDIM; i += blockDim.x) {
    s2[i] = (_Float16)W2[i]; s3[i] = (_Float16)W3[i];
  }
  __syncthreads();
  const int lane = tid & 31;
  const int w = __builtin_amdgcn_readfirstlane(tid >> 5);
  const int hs = lane >> 4, r = lane & 15;
  const int ntiles = (Nv + 15) >> 4;
  const int nwaves = gridDim.x * 4;
  for (int t = blockIdx.x * 4 + w; t < ntiles; t += nwaves) {
    const int nb = t * 16;
    const int node = nb + r;
    // ---- lin1: K=320 over the feature history -----------------------------
    v16h A1[10];
    if (nb + 16 <= Nv) {
#pragma unroll
      for (int kc = 0; kc < 10; ++kc) {
        v16h a;
#pragma unroll
        for (int half = 0; half < 2; ++half) {
          const int kgb = kc * 32 + half * 16 + hs * 8;  // 8 contiguous k's
          const int seg = kgb >> 6, off = kgb & 63;
          load8(xvh + ((size_t)seg * Nv + node) * HDIM + off, 1.f, a, half * 8);
        }
        A1[kc] = a;
      }
    } else {
      const bool valid = node < Nv;
#pragma unroll
      for (int kc = 0; kc < 10; ++kc) {
        v16h a;
#pragma unroll
        for (int tt = 0; tt < 16; ++tt) {
          const int kg = a_k(kc, hs, tt);
          const int seg = kg >> 6, off = kg & 63;
          const float vv = valid ? xvh[((size_t)seg * Nv + node) * HDIM + off] : 0.f;
          a[tt] = (_Float16)vv;
        }
        A1[kc] = a;
      }
    }
#pragma unroll
    for (int nt = 0; nt < 4; ++nt) {
      v8f c = {};
#pragma unroll
      for (int kc = 0; kc < 10; ++kc)
        c = wmma16(A1[kc], bfrag_glb(W1, kc * 32, nt * 16, lane), c);
      const int ch = nt * 16 + r;
      const float bb = b1[ch];
#pragma unroll
      for (int j = 0; j < 8; ++j) obuf[w][j + hs * 8][ch] = fmaxf(c[j] + bb, 0.f);
    }
    __builtin_amdgcn_wave_barrier();
    // ---- lin2 -------------------------------------------------------------
    v16h A2[2];
#pragma unroll
    for (int kc = 0; kc < 2; ++kc) {
      v16h a;
      const int b0 = kc * 32 + hs * 8;
      load8(&obuf[w][r][b0], 1.f, a, 0);
      load8(&obuf[w][r][b0 + 16], 1.f, a, 8);
      A2[kc] = a;
    }
    __builtin_amdgcn_wave_barrier();
#pragma unroll
    for (int nt = 0; nt < 4; ++nt) {
      v8f c = {};
      c = wmma16(A2[0], bfrag_lds(s2, 0, nt * 16, lane), c);
      c = wmma16(A2[1], bfrag_lds(s2, 32, nt * 16, lane), c);
      const int ch = nt * 16 + r;
      const float bb = b2[ch];
#pragma unroll
      for (int j = 0; j < 8; ++j) obuf[w][j + hs * 8][ch] = fmaxf(c[j] + bb, 0.f);
    }
    __builtin_amdgcn_wave_barrier();
    // ---- lin3 -------------------------------------------------------------
    v16h A3[2];
#pragma unroll
    for (int kc = 0; kc < 2; ++kc) {
      v16h a;
      const int b0 = kc * 32 + hs * 8;
      load8(&obuf[w][r][b0], 1.f, a, 0);
      load8(&obuf[w][r][b0 + 16], 1.f, a, 8);
      A3[kc] = a;
    }
    __builtin_amdgcn_wave_barrier();
#pragma unroll
    for (int nt = 0; nt < 4; ++nt) {
      v8f c = {};
      c = wmma16(A3[0], bfrag_lds(s3, 0, nt * 16, lane), c);
      c = wmma16(A3[1], bfrag_lds(s3, 32, nt * 16, lane), c);
      const int ch = nt * 16 + r;
      const float bb = b3[ch];
#pragma unroll
      for (int j = 0; j < 8; ++j) obuf[w][j + hs * 8][ch] = fmaxf(c[j] + bb, 0.f);
    }
    __builtin_amdgcn_wave_barrier();
    // ---- lin4 + log_softmax (2 classes) -----------------------------------
    if (lane < 16) {
      const int nd = nb + lane;
      if (nd < Nv) {
        float z0 = b4[0], z1 = b4[1];
#pragma unroll 8
        for (int k = 0; k < HDIM; ++k) {
          const float x = obuf[w][lane][k];
          z0 += x * W4[k * 2 + 0];
          z1 += x * W4[k * 2 + 1];
        }
        const float mx = fmaxf(z0, z1);
        const float ls = mx + logf(expf(z0 - mx) + expf(z1 - mx));
        out[(size_t)nd * 2 + 0] = z0 - ls;
        out[(size_t)nd * 2 + 1] = z1 - ls;
      }
    }
  }
}

// ---------------------------------------------------------------------------
// Host orchestration
// ---------------------------------------------------------------------------
extern "C" void kernel_launch(void* const* d_in, const int* in_sizes, int n_in,
                              void* d_out, int out_size, void* d_ws,
                              size_t ws_size, hipStream_t stream) {
  (void)n_in; (void)out_size; (void)ws_size;
  const float* vfeat = (const float*)d_in[0];
  const float* cfeat = (const float*)d_in[1];
  const float* efv   = (const float*)d_in[2];
  const float* efc   = (const float*)d_in[3];
  const int*   eiv   = (const int*)d_in[4];
  const int*   eic   = (const int*)d_in[5];
  const float* evW1 = (const float*)d_in[6];
  const float* evb1 = (const float*)d_in[7];
  const float* evW2 = (const float*)d_in[8];
  const float* evb2 = (const float*)d_in[9];
  const float* ecW1 = (const float*)d_in[10];
  const float* ecb1 = (const float*)d_in[11];
  const float* ecW2 = (const float*)d_in[12];
  const float* ecb2 = (const float*)d_in[13];
  const float* lv[9]; const float* lc[9];
  for (int k = 0; k < 9; ++k) { lv[k] = (const float*)d_in[14 + k];
                                lc[k] = (const float*)d_in[23 + k]; }
  const float* lin1W = (const float*)d_in[32];
  const float* lin1b = (const float*)d_in[33];
  const float* lin2W = (const float*)d_in[34];
  const float* lin2b = (const float*)d_in[35];
  const float* lin3W = (const float*)d_in[36];
  const float* lin3b = (const float*)d_in[37];
  const float* lin4W = (const float*)d_in[38];
  const float* lin4b = (const float*)d_in[39];

  const int L = 4;
  const int Nv = in_sizes[0] / 2, Nc = in_sizes[1] / 2;
  const int Ev = in_sizes[2], Ec = in_sizes[3];
  const size_t NvH = (size_t)Nv * HDIM, NcH = (size_t)Nc * HDIM;

  float* ws    = (float*)d_ws;
  float* xvh   = ws;                   // 5 * NvH : x_var history (slot 0 = enc)
  float* xc    = xvh + 5 * NvH;        // NcH
  float* agg   = xc + NcH;             // max(Nv,Nc)*H
  const size_t aggE = (size_t)(Nv > Nc ? Nv : Nc) * HDIM;
  float* cnt_c = agg + aggE;           // Nc
  float* cnt_v = cnt_c + Nc;           // Nv
  float* stats = cnt_v + Nv;           // 128
  float* scsh  = stats + 2 * HDIM;     // 128

  // ---- encoders -----------------------------------------------------------
  encoder_kernel<<<512, 256, 0, stream>>>(vfeat, Nv, evW1, evb1, evW2, evb2, xvh);
  encoder_kernel<<<512, 256, 0, stream>>>(cfeat, Nc, ecW1, ecb1, ecW2, ecb2, xc);

  // ---- in-degree counts (layer-invariant) ---------------------------------
  hipMemsetAsync(cnt_c, 0, (size_t)Nc * sizeof(float), stream);
  hipMemsetAsync(cnt_v, 0, (size_t)Nv * sizeof(float), stream);
  count_kernel<<<(Ev + 255) / 256, 256, 0, stream>>>(eiv, Ev, cnt_c);
  count_kernel<<<(Ec + 255) / 256, 256, 0, stream>>>(eic, Ec, cnt_v);

  const int H2 = HDIM * HDIM;
  for (int i = 0; i < L; ++i) {
    // ---- var -> con (layers_var) ------------------------------------------
    {
      const float *eW1 = lv[0] + i * HDIM, *eb1 = lv[1] + i * HDIM;
      const float *eW2 = lv[2] + i * H2,   *eb2 = lv[3] + i * HDIM;
      const float *g   = lv[4] + i * HDIM, *bt  = lv[5] + i * HDIM;
      const float *lW  = lv[6] + i * H2,   *lb  = lv[7] + i * HDIM;
      const float *rW  = lv[8] + i * H2;
      hipMemsetAsync(stats, 0, 2 * HDIM * sizeof(float), stream);
      hipMemsetAsync(agg, 0, NcH * sizeof(float), stream);
      edge_stats_kernel<<<1024, 256, 0, stream>>>(efv, Ev, eW1, eb1, eW2, eb2, stats);
      bn_finalize_kernel<<<1, 64, 0, stream>>>(stats, g, bt, (float)Ev, scsh);
      edge_message_kernel<<<1024, 256, 0, stream>>>(efv, eiv, Ev, eW1, eb1, eW2,
                                                    eb2, scsh, xvh + i * NvH, agg);
      const int nb = (((Nc + 15) / 16) + 7) / 8;
      node_update_kernel<<<nb, 256, 0, stream>>>(agg, cnt_c, xc, lW, lb, rW, xc, Nc);
    }
    // ---- con -> var (layers_con) ------------------------------------------
    {
      const float *eW1 = lc[0] + i * HDIM, *eb1 = lc[1] + i * HDIM;
      const float *eW2 = lc[2] + i * H2,   *eb2 = lc[3] + i * HDIM;
      const float *g   = lc[4] + i * HDIM, *bt  = lc[5] + i * HDIM;
      const float *lW  = lc[6] + i * H2,   *lb  = lc[7] + i * HDIM;
      const float *rW  = lc[8] + i * H2;
      hipMemsetAsync(stats, 0, 2 * HDIM * sizeof(float), stream);
      hipMemsetAsync(agg, 0, NvH * sizeof(float), stream);
      edge_stats_kernel<<<1024, 256, 0, stream>>>(efc, Ec, eW1, eb1, eW2, eb2, stats);
      bn_finalize_kernel<<<1, 64, 0, stream>>>(stats, g, bt, (float)Ec, scsh);
      edge_message_kernel<<<1024, 256, 0, stream>>>(efc, eic, Ec, eW1, eb1, eW2,
                                                    eb2, scsh, xc, agg);
      const int nb = (((Nv + 15) / 16) + 7) / 8;
      node_update_kernel<<<nb, 256, 0, stream>>>(agg, cnt_v, xvh + i * NvH, lW,
                                                 lb, rW, xvh + (i + 1) * NvH, Nv);
    }
  }

  // ---- final head ---------------------------------------------------------
  const int fb = (((Nv + 15) / 16) + 3) / 4;
  final_mlp_kernel<<<fb, 128, 0, stream>>>(xvh, Nv, lin1W, lin1b, lin2W, lin2b,
                                           lin3W, lin3b, lin4W, lin4b,
                                           (float*)d_out);
}
